// MultiHeadAttention_19413252178500
// MI455X (gfx1250) — compile-verified, matrix operand order checked by reading, not run on hardware
//
#include <hip/hip_runtime.h>

// ---------------- types ----------------
typedef __attribute__((ext_vector_type(16))) _Float16 v16h;
typedef __attribute__((ext_vector_type(8)))  _Float16 v8h;
typedef __attribute__((ext_vector_type(8)))  float    v8f;

#define N_STATE 1024
#define T_Q     256
#define T_CACHE 768
#define T_KV    1024
#define BATCH   8
#define NROWS   2048          // B*T
#define QK_SCALE 0.125f

// ---------------- WMMA helpers ----------------
__device__ __forceinline__ v8f wmma_f16(v16h a, v16h b, v8f c) {
    // D = A(16x32) * B(32x16) + C, f32 accumulate
    return __builtin_amdgcn_wmma_f32_16x16x32_f16(
        false, a, false, b, (short)0, c, false, false);
}

// Load a 16x32 A-style fragment for this lane.
// Lane layout (ISA 7.12.2, 16-bit A 16x32): row = lane&15, kb = (lane>>4)*8,
// halves cover K in [kb,kb+8) and [kb+16,kb+24)  -> two 16-byte loads.
// `p` must point at element [lane&15][0] of the tile (row stride pre-applied).
__device__ __forceinline__ v16h ldfrag(const _Float16* p, int kb) {
    union { v16h v; v8h h[2]; } u;
    u.h[0] = *(const v8h*)(p + kb);
    u.h[1] = *(const v8h*)(p + kb + 16);
    return u.v;
}

// Async global -> LDS copy of 16 bytes per lane (CDNA5 GLOBAL_LOAD_ASYNC_TO_LDS,
// tracked by ASYNCcnt). lds_off = wave-relative LDS byte address (low 32 bits
// of the generic shared pointer).
__device__ __forceinline__ void async_ld_b128(unsigned int lds_off,
                                              const _Float16* g) {
    asm volatile("global_load_async_to_lds_b128 %0, %1, off"
                 :: "v"(lds_off), "v"(g)
                 : "memory");
}

// xor-shuffle across lanes via ds_swizzle (group-of-32 mode: and=0x1f, xor=XM)
template<int XM>
__device__ __forceinline__ float swz_xor(float v) {
    int i = __float_as_int(v);
    i = __builtin_amdgcn_ds_swizzle(i, (XM << 10) | 0x1f);
    return __int_as_float(i);
}

// ---------------- elementwise converts / packs ----------------
__global__ void cvt_f16_kernel(const float* __restrict__ in,
                               _Float16* __restrict__ out, int n) {
    int i = blockIdx.x * 256 + threadIdx.x;
    if (i < n) out[i] = (_Float16)in[i];
}

// W[K=1024][N=1024] f32 -> WT[N][K] f16
__global__ void cvt_wt_kernel(const float* __restrict__ in,
                              _Float16* __restrict__ out) {
    int idx = blockIdx.x * 256 + threadIdx.x;     // over 1M elements
    int k = idx >> 10, n = idx & 1023;
    out[n * 1024 + k] = (_Float16)in[idx];
}

// cache_k [8][768][1024] f32 -> kh [8][1024][1024] f16 (keys 0..767)
__global__ void pack_ck_kernel(const float* __restrict__ ck,
                               _Float16* __restrict__ kh) {
    int idx = blockIdx.x * 256 + threadIdx.x;     // over 8*768*1024
    int c = idx & 1023;
    int s = (idx >> 10) % T_CACHE;
    int b = idx / (T_CACHE * 1024);
    kh[((size_t)(b * T_KV + s)) * 1024 + c] = (_Float16)ck[idx];
}

// cache_v -> vt [b][h][dh][1024 keys] f16 (keys 0..767)
__global__ void pack_cv_kernel(const float* __restrict__ cv,
                               _Float16* __restrict__ vt) {
    int idx = blockIdx.x * 256 + threadIdx.x;
    int c = idx & 1023;
    int s = (idx >> 10) % T_CACHE;
    int b = idx / (T_CACHE * 1024);
    vt[((size_t)((b * 16 + (c >> 6)) * 64 + (c & 63))) * T_KV + s] = (_Float16)cv[idx];
}

// ---------------- tiled WMMA GEMM: C[2048,1024] = A[2048,1024] @ W[1024,1024] ----------------
// A f16 row-major, WT f16 = W^T row-major ([n][k]).
// Block tile 128x128, BK=32; 8 waves in 4(M)x2(N) grid, each wave 32x64 (2x4 16x16 tiles).
// Tiles are DMA'd global->LDS with GLOBAL_LOAD_ASYNC_TO_LDS_B128, double-buffered:
// the DMA of tile k+1 overlaps the 8 WMMAs of tile k.
#define MODE_Q 0
#define MODE_K 1
#define MODE_V 2
#define MODE_O 3

template<int MODE>
__global__ __launch_bounds__(256)
void gemm_wmma_kernel(const _Float16* __restrict__ A,
                      const _Float16* __restrict__ WT,
                      const float* __restrict__ bias,
                      float* __restrict__ out32,
                      _Float16* __restrict__ out16) {
    __shared__ __align__(16) _Float16 As[2][128 * 40];   // [row][40], 32 cols + pad
    __shared__ __align__(16) _Float16 Bs[2][128 * 40];   // [n][40]

    const int tid  = threadIdx.x;
    const int lane = tid & 31;
    const int wid  = tid >> 5;
    const int wm   = wid >> 1;           // 0..3
    const int wn   = wid & 1;            // 0..1
    const int col  = lane & 15;
    const int hi8  = (lane >> 4) * 8;
    const int kb   = hi8;
    const int ablk = blockIdx.y * 128;
    const int bblk = blockIdx.x * 128;

    // each thread owns two 8-half chunks of each tile
    const int q0   = tid,        q1 = tid + 256;          // chunk ids 0..511
    const int row0 = q0 >> 2,    c40 = (q0 & 3) * 8;
    const int row1 = q1 >> 2,    c41 = (q1 & 3) * 8;

    auto issue_tile = [&](int k0, int buf) {
        async_ld_b128((unsigned int)(size_t)&As[buf][row0 * 40 + c40],
                      &A[(size_t)(ablk + row0) * 1024 + k0 + c40]);
        async_ld_b128((unsigned int)(size_t)&Bs[buf][row0 * 40 + c40],
                      &WT[(size_t)(bblk + row0) * 1024 + k0 + c40]);
        async_ld_b128((unsigned int)(size_t)&As[buf][row1 * 40 + c41],
                      &A[(size_t)(ablk + row1) * 1024 + k0 + c41]);
        async_ld_b128((unsigned int)(size_t)&Bs[buf][row1 * 40 + c41],
                      &WT[(size_t)(bblk + row1) * 1024 + k0 + c41]);
    };

    v8f acc[2][4];
#pragma unroll
    for (int mt = 0; mt < 2; ++mt)
#pragma unroll
        for (int nt = 0; nt < 4; ++nt) acc[mt][nt] = (v8f){0,0,0,0,0,0,0,0};

    issue_tile(0, 0);                                   // prologue DMA

#pragma unroll 1
    for (int kk = 0; kk < 32; ++kk) {
        int k0 = kk * 32;
        if (kk + 1 < 32) {
            issue_tile(k0 + 32, (kk + 1) & 1);          // prefetch next tile
            // async loads complete in order: <=4 outstanding => tile kk landed
            asm volatile("s_wait_asynccnt 0x4" ::: "memory");
        } else {
            asm volatile("s_wait_asynccnt 0x0" ::: "memory");
        }
        __syncthreads();

        const _Float16* as = &As[kk & 1][0];
        const _Float16* bs = &Bs[kk & 1][0];
        v16h af[2], bf[4];
#pragma unroll
        for (int mt = 0; mt < 2; ++mt)
            af[mt] = ldfrag(&as[(wm * 32 + mt * 16 + col) * 40], kb);
#pragma unroll
        for (int nt = 0; nt < 4; ++nt)
            bf[nt] = ldfrag(&bs[(wn * 64 + nt * 16 + col) * 40], kb);

#pragma unroll
        for (int mt = 0; mt < 2; ++mt)
#pragma unroll
            for (int nt = 0; nt < 4; ++nt)
                acc[mt][nt] = wmma_f16(af[mt], bf[nt], acc[mt][nt]);
        __syncthreads();   // all waves done reading buf before it is re-DMA'd
    }

    // epilogue (C layout: VGPR r -> row r + 8*(lane>>4), col = lane&15)
#pragma unroll
    for (int mt = 0; mt < 2; ++mt)
#pragma unroll
        for (int nt = 0; nt < 4; ++nt) {
            int rowBase = ablk + wm * 32 + mt * 16 + hi8;
            int c       = bblk + wn * 64 + nt * 16 + col;
#pragma unroll
            for (int r = 0; r < 8; ++r) {
                int g = rowBase + r;                    // 0..2047
                float v = acc[mt][nt][r];
                if constexpr (MODE == MODE_Q) {
                    v = (v + bias[c]) * QK_SCALE;
                    out16[(size_t)g * 1024 + c] = (_Float16)v;
                } else if constexpr (MODE == MODE_K) {
                    out32[(size_t)g * 1024 + c] = v;
                    int b = g >> 8, t = g & 255;
                    out16[((size_t)(b * T_KV + T_CACHE + t)) * 1024 + c] = (_Float16)v;
                } else if constexpr (MODE == MODE_V) {
                    v += bias[c];
                    out32[(size_t)g * 1024 + c] = v;
                    int b = g >> 8, t = g & 255;
                    out16[((size_t)((b * 16 + (c >> 6)) * 64 + (c & 63))) * T_KV
                          + T_CACHE + t] = (_Float16)v;
                } else {
                    v += bias[c];
                    out32[(size_t)g * 1024 + c] = v;
                }
            }
        }
}

// ---------------- flash attention ----------------
// grid = 256 blocks: b(8) x h(16) x qhalf(2); 8 waves x 16 query rows = 128 rows/block.
// Each wave is fully independent (wave-private LDS P buffer, LDS in-order per wave).
__global__ __launch_bounds__(256)
void attn_wmma_kernel(const _Float16* __restrict__ qh,
                      const _Float16* __restrict__ kh,
                      const _Float16* __restrict__ vt,
                      const float* __restrict__ mask,
                      _Float16* __restrict__ ph) {
    __shared__ __align__(16) _Float16 ps[8][16 * 72];

    const int lane  = threadIdx.x & 31;
    const int wid   = threadIdx.x >> 5;
    const int col   = lane & 15;
    const int hi    = lane >> 4;
    const int kb    = hi * 8;
    const int blk   = blockIdx.x;
    const int b     = blk >> 5;
    const int h     = (blk >> 1) & 15;
    const int qbase = (blk & 1) * 128 + wid * 16;

    // Q A-fragments (reused for the whole key loop)
    const _Float16* qrow =
        qh + (size_t)(b * T_Q + qbase + col) * 1024 + h * 64;
    const v16h aq0 = ldfrag(qrow, kb);        // dims 0..31
    const v16h aq1 = ldfrag(qrow + 32, kb);   // dims 32..63

    v8f o[4];
#pragma unroll
    for (int nt = 0; nt < 4; ++nt) o[nt] = (v8f){0,0,0,0,0,0,0,0};
    float rmax[8], rsum[8];
#pragma unroll
    for (int r = 0; r < 8; ++r) { rmax[r] = -3.0e38f; rsum[r] = 0.0f; }

    _Float16* myp = &ps[wid][0];

#pragma unroll 1
    for (int kvc = 0; kvc < T_KV; kvc += 64) {
        // ---- S = Q @ K^T  (16 x 64 scores) ----
        v8f s[4];
#pragma unroll
        for (int g = 0; g < 4; ++g) s[g] = (v8f){0,0,0,0,0,0,0,0};
#pragma unroll
        for (int g = 0; g < 4; ++g) {
            const _Float16* krow =
                kh + (size_t)(b * T_KV + kvc + g * 16 + col) * 1024 + h * 64;
            v16h bk0 = ldfrag(krow, kb);
            v16h bk1 = ldfrag(krow + 32, kb);
            s[g] = wmma_f16(aq0, bk0, s[g]);
            s[g] = wmma_f16(aq1, bk1, s[g]);
        }
        // ---- mask + chunk row-max ----
        float cmax[8];
#pragma unroll
        for (int r = 0; r < 8; ++r) cmax[r] = -3.0e38f;
#pragma unroll
        for (int g = 0; g < 4; ++g)
#pragma unroll
            for (int r = 0; r < 8; ++r) {
                int qr  = qbase + hi * 8 + r;
                int key = kvc + g * 16 + col;
                float v = s[g][r] + mask[(size_t)qr * T_KV + key];
                s[g][r] = v;
                cmax[r] = fmaxf(cmax[r], v);
            }
#pragma unroll
        for (int r = 0; r < 8; ++r) {
            cmax[r] = fmaxf(cmax[r], swz_xor<1>(cmax[r]));
            cmax[r] = fmaxf(cmax[r], swz_xor<2>(cmax[r]));
            cmax[r] = fmaxf(cmax[r], swz_xor<4>(cmax[r]));
            cmax[r] = fmaxf(cmax[r], swz_xor<8>(cmax[r]));
        }
        // ---- online softmax update ----
        float scl[8];
#pragma unroll
        for (int r = 0; r < 8; ++r) {
            float nm = fmaxf(rmax[r], cmax[r]);
            scl[r]   = __expf(rmax[r] - nm);
            rmax[r]  = nm;
        }
#pragma unroll
        for (int nt = 0; nt < 4; ++nt)
#pragma unroll
            for (int r = 0; r < 8; ++r) o[nt][r] *= scl[r];

        float psum[8];
#pragma unroll
        for (int r = 0; r < 8; ++r) psum[r] = 0.0f;
#pragma unroll
        for (int g = 0; g < 4; ++g)
#pragma unroll
            for (int r = 0; r < 8; ++r) {
                float p = __expf(s[g][r] - rmax[r]);
                psum[r] += p;
                myp[(hi * 8 + r) * 72 + g * 16 + col] = (_Float16)p;
            }
#pragma unroll
        for (int r = 0; r < 8; ++r) rsum[r] = rsum[r] * scl[r] + psum[r];

        // ---- O += P @ V (P relaid out via wave-private LDS) ----
        const _Float16* prow = myp + col * 72;
        v16h ap0 = ldfrag(prow, kb);        // keys 0..31 of chunk
        v16h ap1 = ldfrag(prow + 32, kb);   // keys 32..63
#pragma unroll
        for (int nt = 0; nt < 4; ++nt) {
            const _Float16* vrow =
                vt + (size_t)((b * 16 + h) * 64 + nt * 16 + col) * T_KV + kvc;
            v16h bv0 = ldfrag(vrow, kb);
            v16h bv1 = ldfrag(vrow + 32, kb);
            o[nt] = wmma_f16(ap0, bv0, o[nt]);
            o[nt] = wmma_f16(ap1, bv1, o[nt]);
        }
    }

    // ---- finalize: reduce row sums, normalize, write ph (f16) ----
#pragma unroll
    for (int r = 0; r < 8; ++r) {
        rsum[r] += swz_xor<1>(rsum[r]);
        rsum[r] += swz_xor<2>(rsum[r]);
        rsum[r] += swz_xor<4>(rsum[r]);
        rsum[r] += swz_xor<8>(rsum[r]);
    }
#pragma unroll
    for (int nt = 0; nt < 4; ++nt)
#pragma unroll
        for (int r = 0; r < 8; ++r) {
            float v = o[nt][r] / rsum[r];
            int qr  = qbase + hi * 8 + r;
            int c   = h * 64 + nt * 16 + col;
            ph[(size_t)(b * T_Q + qr) * 1024 + c] = (_Float16)v;
        }
}

// ---------------- host launcher ----------------
extern "C" void kernel_launch(void* const* d_in, const int* in_sizes, int n_in,
                              void* d_out, int out_size, void* d_ws, size_t ws_size,
                              hipStream_t stream) {
    const float* x    = (const float*)d_in[0];
    const float* mask = (const float*)d_in[1];
    const float* ck   = (const float*)d_in[2];
    const float* cv   = (const float*)d_in[3];
    const float* Wq   = (const float*)d_in[4];
    const float* bq   = (const float*)d_in[5];
    const float* Wk   = (const float*)d_in[6];
    const float* Wv   = (const float*)d_in[7];
    const float* bv   = (const float*)d_in[8];
    const float* Wo   = (const float*)d_in[9];
    const float* bo   = (const float*)d_in[10];

    float* out  = (float*)d_out;                 // [2048,1024]
    float* outK = out  + (size_t)NROWS * N_STATE;
    float* outV = outK + (size_t)NROWS * N_STATE;

    char* w = (char*)d_ws;
    _Float16* xh  = (_Float16*)w; w += (size_t)NROWS * N_STATE * 2;
    _Float16* WqT = (_Float16*)w; w += (size_t)N_STATE * N_STATE * 2;
    _Float16* WkT = (_Float16*)w; w += (size_t)N_STATE * N_STATE * 2;
    _Float16* WvT = (_Float16*)w; w += (size_t)N_STATE * N_STATE * 2;
    _Float16* WoT = (_Float16*)w; w += (size_t)N_STATE * N_STATE * 2;
    _Float16* qh  = (_Float16*)w; w += (size_t)NROWS * N_STATE * 2;
    _Float16* kh  = (_Float16*)w; w += (size_t)BATCH * T_KV * N_STATE * 2;
    _Float16* vt  = (_Float16*)w; w += (size_t)BATCH * T_KV * N_STATE * 2;
    _Float16* ph  = (_Float16*)w; w += (size_t)NROWS * N_STATE * 2;

    // stage conversions
    cvt_f16_kernel<<<(NROWS * N_STATE) / 256, 256, 0, stream>>>(x, xh, NROWS * N_STATE);
    cvt_wt_kernel<<<(N_STATE * N_STATE) / 256, 256, 0, stream>>>(Wq, WqT);
    cvt_wt_kernel<<<(N_STATE * N_STATE) / 256, 256, 0, stream>>>(Wk, WkT);
    cvt_wt_kernel<<<(N_STATE * N_STATE) / 256, 256, 0, stream>>>(Wv, WvT);
    cvt_wt_kernel<<<(N_STATE * N_STATE) / 256, 256, 0, stream>>>(Wo, WoT);
    pack_ck_kernel<<<(BATCH * T_CACHE * N_STATE) / 256, 256, 0, stream>>>(ck, kh);
    pack_cv_kernel<<<(BATCH * T_CACHE * N_STATE) / 256, 256, 0, stream>>>(cv, vt);

    // projections
    dim3 gg(N_STATE / 128, NROWS / 128);   // (8, 16)
    gemm_wmma_kernel<MODE_Q><<<gg, 256, 0, stream>>>(xh, WqT, bq, nullptr, qh);
    gemm_wmma_kernel<MODE_K><<<gg, 256, 0, stream>>>(xh, WkT, nullptr, outK, kh);
    gemm_wmma_kernel<MODE_V><<<gg, 256, 0, stream>>>(xh, WvT, bv, outV, vt);

    // attention
    attn_wmma_kernel<<<BATCH * 16 * 2, 256, 0, stream>>>(qh, kh, vt, mask, ph);

    // output projection
    gemm_wmma_kernel<MODE_O><<<gg, 256, 0, stream>>>(ph, WoT, bo, out, nullptr);
}
